// DeltaProductBlock_15358803050896
// MI455X (gfx1250) — compile-verified, matrix operand order checked by reading, not run on hardware
//
#include <hip/hip_runtime.h>
#include <hip/hip_bf16.h>
#include <math.h>

// CDNA5 async global->LDS copy path (ASYNCcnt), verified to assemble.
#define USE_ASYNC_STAGE 1

// ---------------- types ----------------
typedef __attribute__((ext_vector_type(16))) __bf16 v16bf;
typedef __attribute__((ext_vector_type(8)))  float  v8f;

union Frag16 { v16bf v; uint4 q[2]; __bf16 e[16]; };

__device__ __forceinline__ __bf16 f2bf(float f) {
  union { float f; unsigned u; } a; a.f = f;
  unsigned r = a.u + 0x7FFFu + ((a.u >> 16) & 1u);   // round-to-nearest-even
  union { unsigned short s; __bf16 b; } c; c.s = (unsigned short)(r >> 16);
  return c.b;
}

// low 32 bits of a generic pointer into LDS == LDS byte address
__device__ __forceinline__ unsigned lds_addr_of(const void* p) {
  return (unsigned)(size_t)p;
}

// Problem constants
#define BB   2
#define TT   2048
#define DD   1024
#define HH   8       // N_HEADS
#define NHH  3       // householder sub-steps
#define HD   128     // HEAD_DIM
#define DV   128     // HEAD_V_DIM
#define BT   (BB*TT) // 4096 tokens

// ---------------- f32 -> bf16 cast (row-major, for activations) -------------
__global__ __launch_bounds__(256) void cast_kernel(const float* __restrict__ in,
                                                   __bf16* __restrict__ out, int n) {
  int i = blockIdx.x * 256 + threadIdx.x;
  if (i < n) out[i] = f2bf(in[i]);
}

// ---------------- f32 [R][C] -> bf16 transposed [C][R] ----------------------
__global__ __launch_bounds__(256) void cast_transpose_kernel(const float* __restrict__ in,
                                                             __bf16* __restrict__ out,
                                                             int R, int C) {
  __shared__ float tile[32][33];
  const int c0 = blockIdx.x * 32, r0 = blockIdx.y * 32;
  const int tx = threadIdx.x & 31, ty = threadIdx.x >> 5;
#pragma unroll
  for (int i = ty; i < 32; i += 8)
    tile[i][tx] = in[(size_t)(r0 + i) * C + (c0 + tx)];
  __syncthreads();
#pragma unroll
  for (int i = ty; i < 32; i += 8)
    out[(size_t)(c0 + i) * R + (r0 + tx)] = f2bf(tile[tx][i]);
}

// ---------------- bf16 WMMA GEMM: C[M,N] = A[M,K] * Bt[N,K]^T ---------------
// Double-buffered LDS tiles pre-swizzled into the CDNA5 16-bit WMMA fragment
// layout; staging via async global->LDS b128 copies overlapped with WMMA.
// All callers use M % 128 == 0 and N % 64 == 0 (no epilogue guards).
#define GM 128
#define GN 64
#define GK 64

__global__ __launch_bounds__(256) void gemm_bf16(const __bf16* __restrict__ A,
                                                 const __bf16* __restrict__ Bt,
                                                 float* __restrict__ C,
                                                 int M, int N, int K) {
  // [buf][ksub][16-row block][lane][2 x uint4 = 16 bf16]
  __shared__ uint4 Af[2][2][8][32][2];   // 32 KB
  __shared__ uint4 Bf[2][2][4][32][2];   // 16 KB

  const int tid  = threadIdx.x;
  const int wave = tid >> 5;
  const int lane = tid & 31;
  const int bm = blockIdx.y * GM;
  const int bn = blockIdx.x * GN;
  const int mb0 = (wave & 3) * 2;
  const int nb0 = (wave >> 2) * 2;

  v8f acc[2][2] = {};

#if !USE_ASYNC_STAGE
  uint4 ra[4], rb[2];
#endif

  auto dstA = [&](int buf, int ch) -> uint4* {
    int r = ch >> 3, c8 = ch & 7, cc = c8 & 3;
    return &Af[buf][c8 >> 2][r >> 4][(r & 15) + ((cc & 1) << 4)][cc >> 1];
  };
  auto dstB = [&](int buf, int ch) -> uint4* {
    int r = ch >> 3, c8 = ch & 7, cc = c8 & 3;
    return &Bf[buf][c8 >> 2][r >> 4][(r & 15) + ((cc & 1) << 4)][cc >> 1];
  };

  auto prefetch_tile = [&](int buf, int k0) {
#if USE_ASYNC_STAGE
#pragma unroll
    for (int it = 0; it < 4; ++it) {
      int ch = it * 256 + tid;
      int r = ch >> 3, c8 = ch & 7;
      const __bf16* src = A + (size_t)(bm + r) * K + k0 + c8 * 8;
      unsigned ldst = lds_addr_of(dstA(buf, ch));
      asm volatile("global_load_async_to_lds_b128 %0, %1, off"
                   :: "v"(ldst), "v"(src) : "memory");
    }
#pragma unroll
    for (int it = 0; it < 2; ++it) {
      int ch = it * 256 + tid;
      int r = ch >> 3, c8 = ch & 7;
      const __bf16* src = Bt + (size_t)(bn + r) * K + k0 + c8 * 8;
      unsigned ldst = lds_addr_of(dstB(buf, ch));
      asm volatile("global_load_async_to_lds_b128 %0, %1, off"
                   :: "v"(ldst), "v"(src) : "memory");
    }
#else
#pragma unroll
    for (int it = 0; it < 4; ++it) {
      int ch = it * 256 + tid; int r = ch >> 3, c8 = ch & 7;
      ra[it] = *(const uint4*)(A + (size_t)(bm + r) * K + k0 + c8 * 8);
    }
#pragma unroll
    for (int it = 0; it < 2; ++it) {
      int ch = it * 256 + tid; int r = ch >> 3, c8 = ch & 7;
      rb[it] = *(const uint4*)(Bt + (size_t)(bn + r) * K + k0 + c8 * 8);
    }
    (void)buf;
#endif
  };

  auto commit_tile = [&](int buf) {
#if USE_ASYNC_STAGE
    asm volatile("s_wait_asynccnt 0x0" ::: "memory");
    (void)buf;
#else
#pragma unroll
    for (int it = 0; it < 4; ++it) *dstA(buf, it * 256 + tid) = ra[it];
#pragma unroll
    for (int it = 0; it < 2; ++it) *dstB(buf, it * 256 + tid) = rb[it];
#endif
  };

  auto compute_tile = [&](int buf) {
#pragma unroll
    for (int ks = 0; ks < 2; ++ks) {
      Frag16 a0, a1, b0, b1;
      a0.q[0] = Af[buf][ks][mb0    ][lane][0]; a0.q[1] = Af[buf][ks][mb0    ][lane][1];
      a1.q[0] = Af[buf][ks][mb0 + 1][lane][0]; a1.q[1] = Af[buf][ks][mb0 + 1][lane][1];
      b0.q[0] = Bf[buf][ks][nb0    ][lane][0]; b0.q[1] = Bf[buf][ks][nb0    ][lane][1];
      b1.q[0] = Bf[buf][ks][nb0 + 1][lane][0]; b1.q[1] = Bf[buf][ks][nb0 + 1][lane][1];
      acc[0][0] = __builtin_amdgcn_wmma_f32_16x16x32_bf16(false, a0.v, false, b0.v, (short)0, acc[0][0], false, false);
      acc[0][1] = __builtin_amdgcn_wmma_f32_16x16x32_bf16(false, a0.v, false, b1.v, (short)0, acc[0][1], false, false);
      acc[1][0] = __builtin_amdgcn_wmma_f32_16x16x32_bf16(false, a1.v, false, b0.v, (short)0, acc[1][0], false, false);
      acc[1][1] = __builtin_amdgcn_wmma_f32_16x16x32_bf16(false, a1.v, false, b1.v, (short)0, acc[1][1], false, false);
    }
  };

  // pipeline: prefetch tile k+1 while computing tile k
  prefetch_tile(0, 0);
  commit_tile(0);
  __syncthreads();
  int buf = 0;
  for (int k0 = 0; k0 < K; k0 += GK) {
    const bool more = (k0 + GK < K);
    if (more) prefetch_tile(buf ^ 1, k0 + GK);
    compute_tile(buf);
    if (more) commit_tile(buf ^ 1);
    __syncthreads();
    buf ^= 1;
  }

  // Epilogue (no guards: exact tile multiples). C/D layout:
  // VGPR r -> M = r + 8*(lane>=16), N = lane&15.
  const int mofs = (lane >> 4) << 3;
  const int gn0  = bn + nb0 * 16 + (lane & 15);
#pragma unroll
  for (int i = 0; i < 2; ++i) {
#pragma unroll
    for (int r = 0; r < 8; ++r) {
      float* crow = C + (size_t)(bm + (mb0 + i) * 16 + mofs + r) * (size_t)N + gn0;
      crow[0]  = acc[i][0][r];
      crow[16] = acc[i][1][r];
    }
  }
}

// ---------------- causal depthwise conv (size 4) + SiLU, 4 ch/thread --------
__global__ __launch_bounds__(256) void conv_silu_kernel(const float* __restrict__ X,
                                                        const float* __restrict__ W,
                                                        float* __restrict__ Y,
                                                        int C, int total4) {
  int q = blockIdx.x * 256 + threadIdx.x;
  if (q >= total4) return;
  int idx = q * 4;
  int c  = idx % C;
  int bt = idx / C;
  int t  = bt & (TT - 1);
  float4 w[4];
#pragma unroll
  for (int j = 0; j < 4; ++j) w[j] = *(const float4*)(W + (size_t)(c + j) * 4);
  float acc[4] = {0.f, 0.f, 0.f, 0.f};
#pragma unroll
  for (int i = 0; i < 4; ++i) {
    int ts = t - 3 + i;
    if (ts >= 0) {
      float4 xv = *(const float4*)(X + (size_t)(bt + i - 3) * C + c);
      const float* xe = &xv.x;
#pragma unroll
      for (int j = 0; j < 4; ++j) acc[j] += xe[j] * (&w[j].x)[i];
    }
  }
  float4 o;
  float* oe = &o.x;
#pragma unroll
  for (int j = 0; j < 4; ++j) oe[j] = acc[j] / (1.f + __expf(-acc[j]));
  *(float4*)(Y + idx) = o;
}

// ---------------- row-wise L2 norm (width 128), in place, with scale --------
__global__ __launch_bounds__(256) void l2norm_kernel(float* __restrict__ X,
                                                     int rows, float scale) {
  int row  = blockIdx.x * 8 + (threadIdx.x >> 5);
  int lane = threadIdx.x & 31;
  if (row >= rows) return;
  float* p = X + (size_t)row * 128;
  float4 v4 = *(const float4*)(p + lane * 4);
  float ss = v4.x * v4.x + v4.y * v4.y + v4.z * v4.z + v4.w * v4.w;
#pragma unroll
  for (int off = 16; off > 0; off >>= 1) ss += __shfl_xor(ss, off, 32);
  float r = rsqrtf(ss + 1e-6f) * scale;
  v4.x *= r; v4.y *= r; v4.z *= r; v4.w *= r;
  *(float4*)(p + lane * 4) = v4;
}

// ---------------- beta / g small projections (N=24 and N=8) ----------------
__global__ __launch_bounds__(256) void small_proj_kernel(const float* __restrict__ X,
    const float* __restrict__ Wb, const float* __restrict__ Wa,
    const float* __restrict__ A_log, const float* __restrict__ dt_bias,
    float* __restrict__ Beta, float* __restrict__ Gout) {
  int token = blockIdx.x * 8 + (threadIdx.x >> 5);
  int lane  = threadIdx.x & 31;
  if (token >= BT) return;
  const float* xr = X + (size_t)token * DD;
  float acc = 0.f;
  if (lane < 24) {
    for (int d = 0; d < DD; ++d) acc += xr[d] * Wb[d * 24 + lane];
    Beta[token * 24 + lane] = 2.f / (1.f + __expf(-acc));  // 2*sigmoid
  } else {
    int hh = lane - 24;
    for (int d = 0; d < DD; ++d) acc += xr[d] * Wa[d * 8 + hh];
    float u  = acc + dt_bias[hh];
    float sp = (u > 20.f) ? u : log1pf(__expf(u));         // softplus
    Gout[token * 8 + hh] = -__expf(A_log[hh]) * sp;
  }
}

// ---------------- gated delta-rule scan ----------------
// 64 blocks = (b,h) x 4 column-chunks of 32. State chunk S[128d x 32e] in
// registers: thread = (dgroup 0..7) x (e_local 0..31), 16 f32 each.
__global__ __launch_bounds__(256) void scan_kernel(const float* __restrict__ Q,
    const float* __restrict__ Kx, const float* __restrict__ V,
    const float* __restrict__ Beta, const float* __restrict__ G,
    float* __restrict__ O) {
  __shared__ __align__(16) float sh_q[HD];
  __shared__ __align__(16) float sh_k[NHH][HD];
  __shared__ __align__(16) float sh_v[NHH][32];
  __shared__ float sh_part[8][32];
  __shared__ float sh_scal[4];   // [0]=g, [1..3]=beta_i

  const int tid = threadIdx.x;
  const int el  = tid & 31;
  const int dg  = tid >> 5;
  const int d0  = dg * 16;
  const int bh  = blockIdx.x >> 2;
  const int ec  = blockIdx.x & 3;
  const int b   = bh >> 3;
  const int h   = bh & 7;
  const int e0  = ec * 32;

  float S[16];
#pragma unroll
  for (int r = 0; r < 16; ++r) S[r] = 0.f;

  for (int t = 0; t < TT; ++t) {
    const size_t tb = (size_t)(b * TT + t);
    // ---- staged b128 loads: q(32 f4), k(96 f4), v(24 f4), 4 scalars ----
    if (tid < 32) {
      *(float4*)&sh_q[tid * 4] = *(const float4*)(Q + tb * 1024 + h * HD + tid * 4);
    } else if (tid < 128) {
      int idx = tid - 32; int i = idx >> 5, sub = idx & 31;
      *(float4*)&sh_k[i][sub * 4] =
          *(const float4*)(Kx + tb * 3072 + i * 1024 + h * HD + sub * 4);
    } else if (tid < 152) {
      int idx = tid - 128; int i = idx >> 3, sub = idx & 7;
      *(float4*)&sh_v[i][sub * 4] =
          *(const float4*)(V + tb * 3072 + i * 1024 + h * DV + e0 + sub * 4);
    } else if (tid < 155) {
      int i = tid - 152;
      sh_scal[1 + i] = Beta[tb * 24 + i * 8 + h];
    } else if (tid == 155) {
      sh_scal[0] = G[tb * 8 + h];
    }
    __syncthreads();

    const float decay = __expf(sh_scal[0]);
#pragma unroll
    for (int r = 0; r < 16; ++r) S[r] *= decay;

    for (int i = 0; i < NHH; ++i) {
      float p = 0.f;
#pragma unroll
      for (int r = 0; r < 16; ++r) p += sh_k[i][d0 + r] * S[r];
      sh_part[dg][el] = p;
      __syncthreads();
      float kS = 0.f;
#pragma unroll
      for (int gdx = 0; gdx < 8; ++gdx) kS += sh_part[gdx][el];
      float coef = sh_scal[1 + i] * (sh_v[i][el] - kS);
#pragma unroll
      for (int r = 0; r < 16; ++r) S[r] += sh_k[i][d0 + r] * coef;
      __syncthreads();
    }

    float p = 0.f;
#pragma unroll
    for (int r = 0; r < 16; ++r) p += sh_q[d0 + r] * S[r];
    sh_part[dg][el] = p;
    __syncthreads();
    if (dg == 0) {
      float oo = 0.f;
#pragma unroll
      for (int gdx = 0; gdx < 8; ++gdx) oo += sh_part[gdx][el];
      O[tb * 1024 + h * DV + e0 + el] = oo;
    }
    __syncthreads();
  }
}

// ---------------- RMS-norm + SiLU gate, write bf16 ----------------
__global__ __launch_bounds__(256) void rmsgate_kernel(const float* __restrict__ O,
    const float* __restrict__ Gt, const float* __restrict__ rmsw,
    __bf16* __restrict__ Out, int rows) {
  int row  = blockIdx.x * 8 + (threadIdx.x >> 5);
  int lane = threadIdx.x & 31;
  if (row >= rows) return;
  const float* op = O  + (size_t)row * 128;
  const float* gp = Gt + (size_t)row * 128;
  float4 v4 = *(const float4*)(op + lane * 4);
  float ss = v4.x * v4.x + v4.y * v4.y + v4.z * v4.z + v4.w * v4.w;
#pragma unroll
  for (int off = 16; off > 0; off >>= 1) ss += __shfl_xor(ss, off, 32);
  float r = rsqrtf(ss * (1.f / 128.f) + 1e-5f);
  float4 g4 = *(const float4*)(gp + lane * 4);
  float4 w4 = *(const float4*)(rmsw + lane * 4);
  const float* ve = &v4.x; const float* ge = &g4.x; const float* we = &w4.x;
#pragma unroll
  for (int i = 0; i < 4; ++i) {
    float gt = ge[i];
    float out = ve[i] * r * we[i] * (gt / (1.f + __expf(-gt)));
    Out[(size_t)row * 128 + lane * 4 + i] = f2bf(out);
  }
}

// ---------------- host launcher ----------------
extern "C" void kernel_launch(void* const* d_in, const int* in_sizes, int n_in,
                              void* d_out, int out_size, void* d_ws, size_t ws_size,
                              hipStream_t stream) {
  const float* x       = (const float*)d_in[0];
  const float* Wq      = (const float*)d_in[3];
  const float* Wk      = (const float*)d_in[4];
  const float* Wv      = (const float*)d_in[5];
  const float* Wb      = (const float*)d_in[6];
  const float* Wa      = (const float*)d_in[7];
  const float* A_log   = (const float*)d_in[8];
  const float* dt_bias = (const float*)d_in[9];
  const float* conv_q  = (const float*)d_in[10];
  const float* conv_k  = (const float*)d_in[11];
  const float* conv_v  = (const float*)d_in[12];
  const float* Wg      = (const float*)d_in[13];
  const float* rms_w   = (const float*)d_in[14];
  const float* Wo      = (const float*)d_in[15];
  float* out = (float*)d_out;

  char* ws = (char*)d_ws;
  size_t off = 0;
  auto alloc = [&](size_t bytes) -> void* {
    void* p = ws + off;
    off += (bytes + 255) & ~(size_t)255;
    return p;
  };

  __bf16* xbf   = (__bf16*)alloc((size_t)BT * DD * 2);
  __bf16* wqT   = (__bf16*)alloc((size_t)1024 * DD * 2);  // [N][K]
  __bf16* wkT   = (__bf16*)alloc((size_t)3072 * DD * 2);
  __bf16* wvT   = (__bf16*)alloc((size_t)3072 * DD * 2);
  __bf16* wgT   = (__bf16*)alloc((size_t)1024 * DD * 2);
  __bf16* woT   = (__bf16*)alloc((size_t)1024 * 1024 * 2);
  float*  qproj = (float*)alloc((size_t)BT * 1024 * 4);   // reused as o_pre
  float*  kproj = (float*)alloc((size_t)BT * 3072 * 4);
  float*  vproj = (float*)alloc((size_t)BT * 3072 * 4);
  float*  gproj = (float*)alloc((size_t)BT * 1024 * 4);
  float*  qc    = (float*)alloc((size_t)BT * 1024 * 4);
  float*  kc    = (float*)alloc((size_t)BT * 3072 * 4);
  float*  vc    = (float*)alloc((size_t)BT * 3072 * 4);
  float*  beta  = (float*)alloc((size_t)BT * 24 * 4);
  float*  gdec  = (float*)alloc((size_t)BT * 8 * 4);
  float*  o_pre = qproj;          // qproj dead after conv
  __bf16* obf   = xbf;            // xbf dead after projections

  dim3 blk(256);

  // 1) activation cast + weight cast-transposes ([K][N] f32 -> [N][K] bf16)
  cast_kernel<<<dim3((BT * DD + 255) / 256), blk, 0, stream>>>(x, xbf, BT * DD);
  cast_transpose_kernel<<<dim3(1024 / 32, DD / 32), blk, 0, stream>>>(Wq, wqT, DD, 1024);
  cast_transpose_kernel<<<dim3(3072 / 32, DD / 32), blk, 0, stream>>>(Wk, wkT, DD, 3072);
  cast_transpose_kernel<<<dim3(3072 / 32, DD / 32), blk, 0, stream>>>(Wv, wvT, DD, 3072);
  cast_transpose_kernel<<<dim3(1024 / 32, DD / 32), blk, 0, stream>>>(Wg, wgT, DD, 1024);
  cast_transpose_kernel<<<dim3(1024 / 32, 1024 / 32), blk, 0, stream>>>(Wo, woT, 1024, 1024);

  // 2) WMMA projections
  gemm_bf16<<<dim3(1024 / GN, BT / GM), blk, 0, stream>>>(xbf, wqT, qproj, BT, 1024, DD);
  gemm_bf16<<<dim3(3072 / GN, BT / GM), blk, 0, stream>>>(xbf, wkT, kproj, BT, 3072, DD);
  gemm_bf16<<<dim3(3072 / GN, BT / GM), blk, 0, stream>>>(xbf, wvT, vproj, BT, 3072, DD);
  gemm_bf16<<<dim3(1024 / GN, BT / GM), blk, 0, stream>>>(xbf, wgT, gproj, BT, 1024, DD);

  // 3) causal conv + SiLU
  conv_silu_kernel<<<dim3((BT * 1024 / 4 + 255) / 256), blk, 0, stream>>>(qproj, conv_q, qc, 1024, BT * 1024 / 4);
  conv_silu_kernel<<<dim3((BT * 3072 / 4 + 255) / 256), blk, 0, stream>>>(kproj, conv_k, kc, 3072, BT * 3072 / 4);
  conv_silu_kernel<<<dim3((BT * 3072 / 4 + 255) / 256), blk, 0, stream>>>(vproj, conv_v, vc, 3072, BT * 3072 / 4);

  // 4) L2 norms (q also scaled by d^-1/2)
  l2norm_kernel<<<dim3((BT * HH + 7) / 8), blk, 0, stream>>>(qc, BT * HH, 0.08838834764831845f);
  l2norm_kernel<<<dim3((BT * HH * NHH + 7) / 8), blk, 0, stream>>>(kc, BT * HH * NHH, 1.0f);

  // 5) beta / g
  small_proj_kernel<<<dim3(BT / 8), blk, 0, stream>>>(x, Wb, Wa, A_log, dt_bias, beta, gdec);

  // 6) recurrent gated delta scan
  scan_kernel<<<dim3(BB * HH * 4), blk, 0, stream>>>(qc, kc, vc, beta, gdec, o_pre);

  // 7) RMS norm + SiLU gate -> bf16
  rmsgate_kernel<<<dim3((BT * HH + 7) / 8), blk, 0, stream>>>(o_pre, gproj, rms_w, obf, BT * HH);

  // 8) output projection -> d_out (f32)
  gemm_bf16<<<dim3(1024 / GN, BT / GM), blk, 0, stream>>>(obf, woT, out, BT, 1024, 1024);

  (void)in_sizes; (void)n_in; (void)out_size; (void)ws_size;
}
